// VRPTWEdgeGNN_8701603741904
// MI455X (gfx1250) — matrix operand order repaired and checked by reading, not compiled, not run on hardware
//
#include <hip/hip_runtime.h>

// ---------------------------------------------------------------------------
// MI455X / gfx1250 fused GNN edge-scoring pipeline.
// wave32, WMMA f32_16x16x32_f16 matrix cores, f16 L2-resident edge features,
// per-wave LDS staging with explicit s_wait_dscnt (CDNA5 split counters).
// ---------------------------------------------------------------------------

typedef _Float16 h16;
typedef __attribute__((ext_vector_type(16))) _Float16 v16h;
typedef __attribute__((ext_vector_type(8)))  float    v8f;

#define NN     25000
#define NE     400000
#define HDIM   128
#define LCONV  4
#define LN_EPS 1e-5f
#define Y1STRIDE 136   // padded LDS row stride in halves (bank-conflict avoidance)

__device__ __forceinline__ v8f wmma16(v16h a, v16h b, v8f c) {
  // D = A(16x32 f16) * B(32x16 f16) + C(16x16 f32)
  return __builtin_amdgcn_wmma_f32_16x16x32_f16(false, a, false, b, (short)0, c,
                                                false, false);
}

__device__ __forceinline__ void lds_fence() {
  // Ensure all prior DS ops retired before subsequent DS ops (in-wave staging).
  asm volatile("s_wait_dscnt 0" ::: "memory");
}

// |d| via a single VALU op, opaque to the optimizer.  Plain fabsf() (and even
// fmaxf(d,-d), which InstCombine re-canonicalizes to llvm.fabs) gets SLP-
// vectorized into a v2f32 fabs bitcast/and pattern that crashes gfx1250 ISel
// in this toolchain.  Inline asm cannot be recognized or vectorized.
__device__ __forceinline__ float absf_hw(float d) {
  float r;
  asm("v_max_num_f32 %0, %1, -%1" : "=v"(r) : "v"(d));
  return r;
}

union AFrag { v16h v; h16 e[16]; };

// ---------------------------------------------------------------------------
// Pack a row-major f32 weight W[K][Ncol] into per-lane-contiguous f16 B
// fragments: frag (nt*ktiles+kt), lane l holds B[kt*32+(l>>4)*8 + j + (j&8)][nt*16+(l&15)]
// at P[(frag*32+l)*16 + j]  -> GEMM loads one contiguous v16h (32B) per lane.
// ---------------------------------------------------------------------------
__global__ void pack_b_kernel(const float* __restrict__ W, h16* __restrict__ P,
                              int K, int Ncol) {
  int gid   = blockIdx.x * blockDim.x + threadIdx.x;
  int ktiles = K >> 5, ntiles = Ncol >> 4;
  int lane  = gid & 31, frag = gid >> 5;
  if (frag >= ktiles * ntiles) return;
  int kt = frag % ktiles, nt = frag / ktiles;
  int n  = nt * 16 + (lane & 15);
  int kb = kt * 32 + (lane >> 4) * 8;
  h16* dst = P + ((size_t)(frag * 32 + lane)) * 16;
#pragma unroll
  for (int j = 0; j < 16; ++j) dst[j] = (h16)W[(size_t)(kb + j + (j & 8)) * Ncol + n];
}

// ---------------------------------------------------------------------------
// Encoder: out = LN(ReLU(x @ W + b)) * g + be.  One wave per row, 4 feats/lane.
// ---------------------------------------------------------------------------
template <int IN_DIM, typename OUT_T>
__global__ void encoder_kernel(const float* __restrict__ x,
                               const float* __restrict__ w,  // [IN_DIM][HDIM]
                               const float* __restrict__ b,
                               const float* __restrict__ g,
                               const float* __restrict__ be,
                               OUT_T* __restrict__ out, int rows) {
  int wave = threadIdx.x >> 5, lane = threadIdx.x & 31;
  int row  = blockIdx.x * 8 + wave;
  if (row >= rows) return;
  const float* xr = x + (size_t)row * IN_DIM;
  float xv[IN_DIM];
#pragma unroll
  for (int d = 0; d < IN_DIM; ++d) xv[d] = xr[d];
  int f0 = lane * 4;
  float y[4];
  float s1 = 0.f, s2 = 0.f;
#pragma unroll
  for (int j = 0; j < 4; ++j) {
    float acc = b[f0 + j];
#pragma unroll
    for (int d = 0; d < IN_DIM; ++d) acc += xv[d] * w[d * HDIM + f0 + j];
    acc = fmaxf(acc, 0.f);
    y[j] = acc;
    s1 += acc;
    s2 += acc * acc;
  }
#pragma unroll
  for (int off = 16; off > 0; off >>= 1) {
    s1 += __shfl_xor(s1, off, 32);
    s2 += __shfl_xor(s2, off, 32);
  }
  float mean = s1 * (1.f / HDIM);
  float var  = s2 * (1.f / HDIM) - mean * mean;
  float rstd = rsqrtf(var + LN_EPS);
  OUT_T* op = out + (size_t)row * HDIM + f0;
#pragma unroll
  for (int j = 0; j < 4; ++j)
    op[j] = (OUT_T)((y[j] - mean) * rstd * g[f0 + j] + be[f0 + j]);
}

// ---------------------------------------------------------------------------
// aggr[dst] += ReLU(h[src] + ea)  -- 4 feats/thread, f32 atomics into L2.
// ---------------------------------------------------------------------------
__global__ void message_kernel(const float* __restrict__ h,
                               const h16* __restrict__ ea,
                               const int* __restrict__ src,
                               const int* __restrict__ dst,
                               float* __restrict__ aggr) {
  unsigned tid = blockIdx.x * blockDim.x + threadIdx.x;
  unsigned e = tid >> 5;
  if (e >= NE) return;
  int f = (tid & 31) * 4;
  int s = src[e], d = dst[e];
  const float4 hs = *(const float4*)(h + (size_t)s * HDIM + f);
  const h16* ep = ea + (size_t)e * HDIM + f;
  float m0 = fmaxf(hs.x + (float)ep[0], 0.f);
  float m1 = fmaxf(hs.y + (float)ep[1], 0.f);
  float m2 = fmaxf(hs.z + (float)ep[2], 0.f);
  float m3 = fmaxf(hs.w + (float)ep[3], 0.f);
  float* ap = aggr + (size_t)d * HDIM + f;
  atomicAdd(ap + 0, m0);
  atomicAdd(ap + 1, m1);
  atomicAdd(ap + 2, m2);
  atomicAdd(ap + 3, m3);
}

__global__ void zero_kernel(float4* __restrict__ p, int n4) {
  int i = blockIdx.x * blockDim.x + threadIdx.x;
  if (i < n4) p[i] = make_float4(0.f, 0.f, 0.f, 0.f);
}

// ---------------------------------------------------------------------------
// Fused GINE conv layer:  z = h + aggr
//   y1 = ReLU(z @ W1 + b1)           (32 WMMA, stage to LDS f16)
//   z2 = y1 @ W2 + b2                (32 WMMA)
//   h  = ReLU(LN(z2)*g + bn) + h     (in-register cross-lane LayerNorm)
// One 16-node tile per wave; 8 independent waves per block (no block barrier).
// ---------------------------------------------------------------------------
__global__ void conv_gemm_kernel(float* __restrict__ h,
                                 const float* __restrict__ aggr,
                                 const h16* __restrict__ pw1,
                                 const h16* __restrict__ pw2,
                                 const float* __restrict__ b1,
                                 const float* __restrict__ b2,
                                 const float* __restrict__ g,
                                 const float* __restrict__ bn,
                                 int numRows) {
  __shared__ h16 ysm[8][16 * Y1STRIDE];  // 34,816 B
  int wave = threadIdx.x >> 5, lane = threadIdx.x & 31;
  int tile = blockIdx.x * 8 + wave;
  int numTiles = (numRows + 15) >> 4;
  if (tile >= numTiles) return;
  int m = lane & 15, hi = lane >> 4;
  int rowc = min(tile * 16 + m, numRows - 1);
  h16* y1 = &ysm[wave][0];

  // ---- build A fragments for z = h + aggr (K = 128 -> 4 fragments) ----
  v16h afrag[4];
#pragma unroll
  for (int kt = 0; kt < 4; ++kt) {
    int k0 = kt * 32 + hi * 8;
    const float* hp = h + (size_t)rowc * HDIM + k0;
    const float* ap = aggr + (size_t)rowc * HDIM + k0;
    AFrag u;
#pragma unroll
    for (int j = 0; j < 8; ++j) {
      u.e[j]     = (h16)(hp[j] + ap[j]);
      u.e[8 + j] = (h16)(hp[16 + j] + ap[16 + j]);
    }
    afrag[kt] = u.v;
  }

  // ---- stage 1: y1 = ReLU(z @ W1 + b1) -> LDS f16 ----
#pragma unroll
  for (int nt = 0; nt < 8; ++nt) {
    v8f acc = {};
#pragma unroll
    for (int kt = 0; kt < 4; ++kt) {
      v16h bf = *(const v16h*)(pw1 + ((size_t)((nt * 4 + kt) * 32 + lane)) * 16);
      acc = wmma16(afrag[kt], bf, acc);
    }
    int n = nt * 16 + m;
    float bias = b1[n];
#pragma unroll
    for (int v = 0; v < 8; ++v)
      y1[(v + 8 * hi) * Y1STRIDE + n] = (h16)fmaxf(acc[v] + bias, 0.f);
  }
  lds_fence();

  // ---- stage 2: z2 = y1 @ W2 ----
  v8f acc2[8];
#pragma unroll
  for (int nt = 0; nt < 8; ++nt) acc2[nt] = {};
#pragma unroll
  for (int kt = 0; kt < 4; ++kt) {
    int k0 = kt * 32 + hi * 8;
    const h16* yp = &y1[m * Y1STRIDE + k0];
    AFrag u;
#pragma unroll
    for (int j = 0; j < 8; ++j) {
      u.e[j]     = yp[j];
      u.e[8 + j] = yp[16 + j];
    }
#pragma unroll
    for (int nt = 0; nt < 8; ++nt) {
      v16h bf = *(const v16h*)(pw2 + ((size_t)((nt * 4 + kt) * 32 + lane)) * 16);
      acc2[nt] = wmma16(u.v, bf, acc2[nt]);
    }
  }

  // ---- in-register LayerNorm + ReLU + residual, write h in place ----
  float s1v[8], s2v[8];
#pragma unroll
  for (int v = 0; v < 8; ++v) { s1v[v] = 0.f; s2v[v] = 0.f; }
#pragma unroll
  for (int nt = 0; nt < 8; ++nt) {
    float bias = b2[nt * 16 + m];
#pragma unroll
    for (int v = 0; v < 8; ++v) {
      float val = acc2[nt][v] + bias;
      acc2[nt][v] = val;
      s1v[v] += val;
      s2v[v] += val * val;
    }
  }
#pragma unroll
  for (int off = 1; off < 16; off <<= 1) {
#pragma unroll
    for (int v = 0; v < 8; ++v) {
      s1v[v] += __shfl_xor(s1v[v], off, 32);
      s2v[v] += __shfl_xor(s2v[v], off, 32);
    }
  }
  float mean[8], rstd[8];
#pragma unroll
  for (int v = 0; v < 8; ++v) {
    mean[v] = s1v[v] * (1.f / HDIM);
    float var = s2v[v] * (1.f / HDIM) - mean[v] * mean[v];
    rstd[v] = rsqrtf(var + LN_EPS);
  }
#pragma unroll
  for (int nt = 0; nt < 8; ++nt) {
    int n = nt * 16 + m;
    float gg = g[n], bb = bn[n];
#pragma unroll
    for (int v = 0; v < 8; ++v) {
      int row = tile * 16 + v + 8 * hi;
      if (row < numRows) {
        float* hp = h + (size_t)row * HDIM + n;
        float o = fmaxf((acc2[nt][v] - mean[v]) * rstd[v] * gg + bb, 0.f) + *hp;
        *hp = o;
      }
    }
  }
}

// ---------------------------------------------------------------------------
// Fused edge-scoring MLP over feat = [h[src] | h[dst] | |h[src]-h[dst]| | ea]:
//   s = ReLU(feat @ W1 + b1)   (512-K: 128 WMMA, A built in-register)
//   s = ReLU(s @ W2 + b2)      (16 WMMA via LDS-staged f16)
//   out = s @ w3 + b3          (in-register dot + cross-lane reduce)
// One 16-edge tile per wave.
// ---------------------------------------------------------------------------
__global__ void edge_mlp_kernel(const float* __restrict__ h,
                                const h16* __restrict__ ea,
                                const int* __restrict__ srcI,
                                const int* __restrict__ dstI,
                                const h16* __restrict__ pm1,  // [512x128] packed
                                const float* __restrict__ b1,
                                const h16* __restrict__ pm2,  // [128x64] packed
                                const float* __restrict__ b2,
                                const float* __restrict__ w3,  // [64]
                                const float* __restrict__ b3,  // [1]
                                float* __restrict__ out) {
  __shared__ h16 ysm[8][16 * Y1STRIDE];
  int wave = threadIdx.x >> 5, lane = threadIdx.x & 31;
  int tile = blockIdx.x * 8 + wave;  // NE/16 tiles exactly
  int m = lane & 15, hi = lane >> 4;
  int e = tile * 16 + m;
  int s = srcI[e], d = dstI[e];
  const float* hs = h + (size_t)s * HDIM;
  const float* hd = h + (size_t)d * HDIM;
  const h16* ep = ea + (size_t)e * HDIM;
  h16* y1 = &ysm[wave][0];

  // ---- stage 1: 8 column tiles, K = 512 (16 fragments) ----
  v8f acc[8];
#pragma unroll
  for (int nt = 0; nt < 8; ++nt) acc[nt] = {};
#pragma unroll
  for (int kt = 0; kt < 16; ++kt) {
    int seg = kt >> 2;                 // 0:src 1:dst 2:|diff| 3:ea
    int kk = (kt & 3) * 32 + hi * 8;   // offset within 128-wide segment
    AFrag u;
    if (seg == 0) {
#pragma unroll
      for (int j = 0; j < 8; ++j) {
        u.e[j]     = (h16)hs[kk + j];
        u.e[8 + j] = (h16)hs[kk + 16 + j];
      }
    } else if (seg == 1) {
#pragma unroll
      for (int j = 0; j < 8; ++j) {
        u.e[j]     = (h16)hd[kk + j];
        u.e[8 + j] = (h16)hd[kk + 16 + j];
      }
    } else if (seg == 2) {
#pragma unroll
      for (int j = 0; j < 8; ++j) {
        u.e[j]     = (h16)absf_hw(hs[kk + j] - hd[kk + j]);
        u.e[8 + j] = (h16)absf_hw(hs[kk + 16 + j] - hd[kk + 16 + j]);
      }
    } else {
#pragma unroll
      for (int j = 0; j < 8; ++j) {
        u.e[j]     = ep[kk + j];
        u.e[8 + j] = ep[kk + 16 + j];
      }
    }
#pragma unroll
    for (int nt = 0; nt < 8; ++nt) {
      v16h bf = *(const v16h*)(pm1 + ((size_t)((nt * 16 + kt) * 32 + lane)) * 16);
      acc[nt] = wmma16(u.v, bf, acc[nt]);
    }
  }
#pragma unroll
  for (int nt = 0; nt < 8; ++nt) {
    int n = nt * 16 + m;
    float bias = b1[n];
#pragma unroll
    for (int v = 0; v < 8; ++v)
      y1[(v + 8 * hi) * Y1STRIDE + n] = (h16)fmaxf(acc[nt][v] + bias, 0.f);
  }
  lds_fence();

  // ---- stage 2: [16,128] @ [128,64] ----
  v8f a2[4];
#pragma unroll
  for (int t = 0; t < 4; ++t) a2[t] = {};
#pragma unroll
  for (int kt = 0; kt < 4; ++kt) {
    int k0 = kt * 32 + hi * 8;
    const h16* yp = &y1[m * Y1STRIDE + k0];
    AFrag u;
#pragma unroll
    for (int j = 0; j < 8; ++j) {
      u.e[j]     = yp[j];
      u.e[8 + j] = yp[16 + j];
    }
#pragma unroll
    for (int t = 0; t < 4; ++t) {
      v16h bf = *(const v16h*)(pm2 + ((size_t)((t * 4 + kt) * 32 + lane)) * 16);
      a2[t] = wmma16(u.v, bf, a2[t]);
    }
  }

  // ---- stage 3: ReLU(+b2) then dot with w3, cross-lane reduce ----
  float part[8];
#pragma unroll
  for (int v = 0; v < 8; ++v) part[v] = 0.f;
#pragma unroll
  for (int t = 0; t < 4; ++t) {
    int n = t * 16 + m;
    float bias = b2[n], wv = w3[n];
#pragma unroll
    for (int v = 0; v < 8; ++v)
      part[v] += fmaxf(a2[t][v] + bias, 0.f) * wv;
  }
#pragma unroll
  for (int off = 1; off < 16; off <<= 1)
#pragma unroll
    for (int v = 0; v < 8; ++v) part[v] += __shfl_xor(part[v], off, 32);
  if (m == 0) {
    float bb = b3[0];
#pragma unroll
    for (int v = 0; v < 8; ++v) out[tile * 16 + v + 8 * hi] = part[v] + bb;
  }
}

// ---------------------------------------------------------------------------
// Host launcher
// ---------------------------------------------------------------------------
extern "C" void kernel_launch(void* const* d_in, const int* in_sizes, int n_in,
                              void* d_out, int out_size, void* d_ws, size_t ws_size,
                              hipStream_t stream) {
  const float* x         = (const float*)d_in[0];
  const float* edge_attr = (const float*)d_in[1];
  const int*   eidx      = (const int*)d_in[2];
  const float* ne_w = (const float*)d_in[3];
  const float* ne_b = (const float*)d_in[4];
  const float* ne_g = (const float*)d_in[5];
  const float* ne_be = (const float*)d_in[6];
  const float* ee_w = (const float*)d_in[7];
  const float* ee_b = (const float*)d_in[8];
  const float* ee_g = (const float*)d_in[9];
  const float* ee_be = (const float*)d_in[10];
  const float* conv_w1 = (const float*)d_in[11];
  const float* conv_b1 = (const float*)d_in[12];
  const float* conv_w2 = (const float*)d_in[13];
  const float* conv_b2 = (const float*)d_in[14];
  const float* norm_g = (const float*)d_in[15];
  const float* norm_b = (const float*)d_in[16];
  const float* mlp_w1 = (const float*)d_in[17];
  const float* mlp_b1 = (const float*)d_in[18];
  const float* mlp_w2 = (const float*)d_in[19];
  const float* mlp_b2 = (const float*)d_in[20];
  const float* mlp_w3 = (const float*)d_in[21];
  const float* mlp_b3 = (const float*)d_in[22];

  // ---- workspace carve (all 256B aligned) ----
  char* ws = (char*)d_ws;
  float* h    = (float*)ws;  ws += (size_t)NN * HDIM * sizeof(float);   // 12.8 MB
  float* aggr = (float*)ws;  ws += (size_t)NN * HDIM * sizeof(float);   // 12.8 MB
  h16*  eaF   = (h16*)ws;    ws += (size_t)NE * HDIM * sizeof(h16);     // 102.4 MB
  h16*  pw1   = (h16*)ws;    ws += (size_t)LCONV * 16384 * sizeof(h16);
  h16*  pw2   = (h16*)ws;    ws += (size_t)LCONV * 16384 * sizeof(h16);
  h16*  pm1   = (h16*)ws;    ws += (size_t)65536 * sizeof(h16);
  h16*  pm2   = (h16*)ws;    ws += (size_t)8192 * sizeof(h16);

  const int* srcI = eidx;
  const int* dstI = eidx + NE;

  // ---- pack WMMA B operands (tiny; re-done every call for determinism) ----
  for (int l = 0; l < LCONV; ++l) {
    pack_b_kernel<<<4, 256, 0, stream>>>(conv_w1 + (size_t)l * HDIM * HDIM,
                                         pw1 + (size_t)l * 16384, HDIM, HDIM);
    pack_b_kernel<<<4, 256, 0, stream>>>(conv_w2 + (size_t)l * HDIM * HDIM,
                                         pw2 + (size_t)l * 16384, HDIM, HDIM);
  }
  pack_b_kernel<<<16, 256, 0, stream>>>(mlp_w1, pm1, 4 * HDIM, HDIM);
  pack_b_kernel<<<2, 256, 0, stream>>>(mlp_w2, pm2, HDIM, 64);

  // ---- encoders ----
  encoder_kernel<8, float><<<(NN + 7) / 8, 256, 0, stream>>>(
      x, ne_w, ne_b, ne_g, ne_be, h, NN);
  encoder_kernel<4, h16><<<(NE + 7) / 8, 256, 0, stream>>>(
      edge_attr, ee_w, ee_b, ee_g, ee_be, eaF, NE);

  // ---- L GINEConv layers ----
  int numTiles = (NN + 15) / 16;                       // 1563
  int convBlocks = (numTiles + 7) / 8;                 // 196
  for (int l = 0; l < LCONV; ++l) {
    zero_kernel<<<(NN * HDIM / 4 + 255) / 256, 256, 0, stream>>>(
        (float4*)aggr, NN * HDIM / 4);
    message_kernel<<<(NE * 32) / 256, 256, 0, stream>>>(h, eaF, srcI, dstI, aggr);
    conv_gemm_kernel<<<convBlocks, 256, 0, stream>>>(
        h, aggr, pw1 + (size_t)l * 16384, pw2 + (size_t)l * 16384,
        conv_b1 + l * HDIM, conv_b2 + l * HDIM,
        norm_g + l * HDIM, norm_b + l * HDIM, NN);
  }

  // ---- final edge MLP ----
  edge_mlp_kernel<<<(NE / 16) / 8, 256, 0, stream>>>(
      h, eaF, srcI, dstI, pm1, mlp_b1, pm2, mlp_b2, mlp_w3, mlp_b3,
      (float*)d_out);
}
